// ReacFormerSO2Block_38732015075516
// MI455X (gfx1250) — compile-verified, matrix operand order checked by reading, not run on hardware
//
#include <hip/hip_runtime.h>
#include <math.h>

#define NN 512
#define DD 128
#define HH 8

typedef __attribute__((ext_vector_type(16))) __bf16 v16bf;
typedef __attribute__((ext_vector_type(8)))  float  v8f;

__device__ __forceinline__ v8f wmma_bf16(v16bf a, v16bf b, v8f c) {
  return __builtin_amdgcn_wmma_f32_16x16x32_bf16(false, a, false, b, (short)0, c, false, false);
}
__device__ __forceinline__ float gelu_t(float x) {
  return 0.5f * x * (1.f + tanhf(0.7978845608028654f * (x + 0.044715f * x * x * x)));
}
__device__ __forceinline__ float sigm(float x) { return 1.f / (1.f + __expf(-x)); }

// ---------------------------------------------------------------------------
// K0: swizzle a (K x C) f32 weight matrix into per-lane WMMA B-fragment layout
// frag[tile][lane][h] (bf16), tile = kc*(Cpad/16)+t ; k = kc*32 + kmap(lane,h)
// kmap: h<8 -> h + 8*(lane>=16) ; h>=8 -> h+8 + 8*(lane>=16)
// ---------------------------------------------------------------------------
__global__ void k_swz(const float* __restrict__ W, __bf16* __restrict__ out,
                      int K, int C, int Cpad) {
  int idx = blockIdx.x * 256 + threadIdx.x;
  int total = (K >> 5) * (Cpad >> 4) * 512;
  if (idx >= total) return;
  int h = idx & 15, lane = (idx >> 4) & 31, tile = idx >> 9;
  int nt = Cpad >> 4;
  int kc = tile / nt, t = tile - kc * nt;
  int kk = (h & 7) + ((h >> 3) << 4) + ((lane >> 4) << 3);
  int k = kc * 32 + kk;
  int col = t * 16 + (lane & 15);
  out[idx] = (__bf16)((col < C) ? W[k * C + col] : 0.f);
}

// ---------------------------------------------------------------------------
// K1: per-node preprocessing: LN1 -> q,k,v ; phi = silu(s@b_w1+b)@b_w2+b ; norms
// ---------------------------------------------------------------------------
__global__ __launch_bounds__(128) void k_node(
    const float* __restrict__ node_s, const float* __restrict__ node_v,
    const float* __restrict__ node_t,
    const float* __restrict__ ln1_g, const float* __restrict__ ln1_b,
    const float* __restrict__ wq, const float* __restrict__ bq,
    const float* __restrict__ wk, const float* __restrict__ bk,
    const float* __restrict__ wv, const float* __restrict__ bv,
    const float* __restrict__ b_w1, const float* __restrict__ b_b1,
    const float* __restrict__ b_w2, const float* __restrict__ b_b2,
    float* __restrict__ qo, float* __restrict__ ko, float* __restrict__ vo,
    float* __restrict__ phio, float* __restrict__ vno, float* __restrict__ tno) {
  __shared__ float s_s[DD], h_s[DD], u_s[DD], red[2];
  const int i = blockIdx.x, d = threadIdx.x;
  s_s[d] = node_s[(size_t)i * DD + d];
  __syncthreads();
  if (d == 0) { red[0] = 0.f; red[1] = 0.f; }
  __syncthreads();
  atomicAdd(&red[0], s_s[d]);
  atomicAdd(&red[1], s_s[d] * s_s[d]);
  __syncthreads();
  float mean = red[0] * (1.f / DD), var = red[1] * (1.f / DD) - mean * mean;
  float rstd = rsqrtf(var + 1e-5f);
  h_s[d] = (s_s[d] - mean) * rstd * ln1_g[d] + ln1_b[d];
  {
    float t2 = b_b1[d];
    for (int k2 = 0; k2 < DD; ++k2) t2 += s_s[k2] * b_w1[(size_t)k2 * DD + d];
    u_s[d] = t2 * sigm(t2);  // silu
  }
  __syncthreads();
  float sq = bq[d], sk = bk[d], sv = bv[d];
  for (int k2 = 0; k2 < DD; ++k2) {
    float hv = h_s[k2];
    sq += hv * wq[(size_t)k2 * DD + d];
    sk += hv * wk[(size_t)k2 * DD + d];
    sv += hv * wv[(size_t)k2 * DD + d];
  }
  qo[(size_t)i * DD + d] = sq;
  ko[(size_t)i * DD + d] = sk;
  vo[(size_t)i * DD + d] = sv;
  for (int b5 = 0; b5 < 5; ++b5) {
    int col = b5 * DD + d;
    float s = b_b2[col];
    for (int k2 = 0; k2 < DD; ++k2) s += u_s[k2] * b_w2[(size_t)k2 * 640 + col];
    phio[(size_t)i * 640 + col] = s;
  }
  {
    const float* vp = node_v + (size_t)i * 3 * DD + d;
    float a = vp[0], b = vp[DD], c = vp[2 * DD];
    vno[(size_t)i * DD + d] = sqrtf(a * a + b * b + c * c);
  }
  {
    const float* tp = node_t + (size_t)i * 5 * DD + d;
    float s = 0.f;
#pragma unroll
    for (int c = 0; c < 5; ++c) { float a = tp[c * DD]; s += a * a; }
    tno[(size_t)i * DD + d] = sqrtf(s);
  }
}

// ---------------------------------------------------------------------------
// K2: fused pair kernel. Block = row i (8 waves). For each 16-j tile:
//   F = rbf@b_wf + b_bf (40 WMMA tiles), G = rbf@c_wf + c_bf (8 WMMA tiles),
//   then masked j-contractions accumulated into LDS (no 671MB intermediate).
// ---------------------------------------------------------------------------
__global__ __launch_bounds__(256) void k_pair(
    const float* __restrict__ rbf, const float* __restrict__ rhat,
    const unsigned char* __restrict__ amask, const float* __restrict__ phi,
    const float* __restrict__ node_v, const float* __restrict__ node_t,
    const __bf16* __restrict__ fr_bwf, const __bf16* __restrict__ fr_cwf,
    const float* __restrict__ b_bf, const float* __restrict__ c_bf,
    float* __restrict__ o_sb, float* __restrict__ o_v, float* __restrict__ o_t,
    float* __restrict__ o_A) {
  __shared__ float acc_sb[DD];
  __shared__ float acc_v[3 * DD];
  __shared__ float acc_t[5 * DD];
  __shared__ float acc_A[16 * DD];
  __shared__ float Ys[8][16][16];
  __shared__ float rhs[8][16][4];
  __shared__ float mks[8][16];

  const int i = blockIdx.x;
  const int tid = threadIdx.x;
  const int lane = tid & 31;
  const int wave = tid >> 5;
  const int mlo = lane & 15;
  const int khi = (lane >> 4) << 3;
  const int mbase = khi;

  for (int t = tid; t < 3200; t += 256) {
    if (t < DD) acc_sb[t] = 0.f;
    else if (t < 4 * DD) acc_v[t - DD] = 0.f;
    else if (t < 9 * DD) acc_t[t - 4 * DD] = 0.f;
    else acc_A[t - 9 * DD] = 0.f;
  }
  __syncthreads();

  for (int it = 0; it < 4; ++it) {
    const int tj = wave + it * 8;
    const int j0 = tj * 16;
    if (lane < 16) {
      const int j = j0 + lane;
      const float* rp = rhat + ((size_t)i * NN + j) * 3;
      float x = rp[0], y = rp[1], z = rp[2];
      rhs[wave][lane][0] = x; rhs[wave][lane][1] = y; rhs[wave][lane][2] = z;
      float xy = x * y, yz = y * z, xz = x * z, x2 = x * x, y2 = y * y, z2 = z * z;
      float* Yp = Ys[wave][lane];
      Yp[0] = 0.28209479177387814f;
      Yp[1] = 0.4886025119029199f * x;
      Yp[2] = 0.4886025119029199f * y;
      Yp[3] = 0.4886025119029199f * z;
      Yp[4] = 1.0925484305920792f * xy;
      Yp[5] = 1.0925484305920792f * yz;
      Yp[6] = 0.31539156525252005f * (3.f * z2 - 1.f);
      Yp[7] = 1.0925484305920792f * xz;
      Yp[8] = 0.5462742152960396f * (x2 - y2);
      Yp[9] = 0.5900435899266435f * y * (3.f * x2 - y2);
      Yp[10] = 2.890611442640554f * xy * z;
      Yp[11] = 0.4570457994644658f * y * (5.f * z2 - 1.f);
      Yp[12] = 0.3731763325901154f * z * (5.f * z2 - 3.f);
      Yp[13] = 0.4570457994644658f * x * (5.f * z2 - 1.f);
      Yp[14] = 1.445305721320277f * z * (x2 - y2);
      Yp[15] = 0.5900435899266435f * x * (3.f * x2 - y2);
      mks[wave][lane] = amask[(size_t)i * NN + j] ? 1.f : 0.f;
    }
    __syncthreads();

    // A fragment: rbf[i, j0+m, k] in ISA 16-bit A layout
    v16bf afr;
    {
      const float* r = rbf + (((size_t)i * NN + j0 + mlo) * 32) + khi;
#pragma unroll
      for (int h2 = 0; h2 < 8; ++h2) {
        afr[h2] = (__bf16)r[h2];
        afr[h2 + 8] = (__bf16)r[16 + h2];
      }
    }

#pragma unroll 1
    for (int tg = 0; tg < 48; ++tg) {
      const int isG = (tg >= 40);
      const __bf16* bp = isG ? (fr_cwf + (((size_t)(tg - 40) * 32 + lane) * 16))
                             : (fr_bwf + (((size_t)tg * 32 + lane) * 16));
      v16bf bfr = *(const v16bf*)bp;
      const int colg = (isG ? (tg - 40) : tg) * 16 + mlo;
      const float bias = isG ? c_bf[colg] : b_bf[colg];
      v8f c;
#pragma unroll
      for (int r8 = 0; r8 < 8; ++r8) c[r8] = bias;
      c = wmma_bf16(afr, bfr, c);
      const int d = colg & (DD - 1);
      const int blk = isG ? 5 : (tg >> 3);
      if (blk == 0) {  // Fs * ps
        float s = 0.f;
#pragma unroll
        for (int r8 = 0; r8 < 8; ++r8) {
          int m = r8 + mbase; int j = j0 + m;
          s += c[r8] * mks[wave][m] * phi[(size_t)j * 640 + d];
        }
        atomicAdd(&acc_sb[d], s);
      } else if (blk == 1) {  // Fvv * pvv * node_v
        float s0 = 0.f, s1 = 0.f, s2 = 0.f;
#pragma unroll
        for (int r8 = 0; r8 < 8; ++r8) {
          int m = r8 + mbase; int j = j0 + m;
          float w = c[r8] * mks[wave][m] * phi[(size_t)j * 640 + DD + d];
          const float* vp = node_v + (size_t)j * 3 * DD + d;
          s0 += w * vp[0]; s1 += w * vp[DD]; s2 += w * vp[2 * DD];
        }
        atomicAdd(&acc_v[d], s0);
        atomicAdd(&acc_v[DD + d], s1);
        atomicAdd(&acc_v[2 * DD + d], s2);
      } else if (blk == 2) {  // Fvs * pvs * r_hat
        float s0 = 0.f, s1 = 0.f, s2 = 0.f;
#pragma unroll
        for (int r8 = 0; r8 < 8; ++r8) {
          int m = r8 + mbase; int j = j0 + m;
          float w = c[r8] * mks[wave][m] * phi[(size_t)j * 640 + 2 * DD + d];
          s0 += w * rhs[wave][m][0];
          s1 += w * rhs[wave][m][1];
          s2 += w * rhs[wave][m][2];
        }
        atomicAdd(&acc_v[d], s0);
        atomicAdd(&acc_v[DD + d], s1);
        atomicAdd(&acc_v[2 * DD + d], s2);
      } else if (blk == 3) {  // Ftt * ptt * node_t
        float s[5] = {0.f, 0.f, 0.f, 0.f, 0.f};
#pragma unroll
        for (int r8 = 0; r8 < 8; ++r8) {
          int m = r8 + mbase; int j = j0 + m;
          float w = c[r8] * mks[wave][m] * phi[(size_t)j * 640 + 3 * DD + d];
          const float* tp = node_t + (size_t)j * 5 * DD + d;
#pragma unroll
          for (int c5 = 0; c5 < 5; ++c5) s[c5] += w * tp[c5 * DD];
        }
#pragma unroll
        for (int c5 = 0; c5 < 5; ++c5) atomicAdd(&acc_t[c5 * DD + d], s[c5]);
      } else if (blk == 4) {  // Fts * pts * T2(Y[4..8])
        float s[5] = {0.f, 0.f, 0.f, 0.f, 0.f};
#pragma unroll
        for (int r8 = 0; r8 < 8; ++r8) {
          int m = r8 + mbase; int j = j0 + m;
          float w = c[r8] * mks[wave][m] * phi[(size_t)j * 640 + 4 * DD + d];
#pragma unroll
          for (int c5 = 0; c5 < 5; ++c5) s[c5] += w * Ys[wave][m][4 + c5];
        }
#pragma unroll
        for (int c5 = 0; c5 < 5; ++c5) atomicAdd(&acc_t[c5 * DD + d], s[c5]);
      } else {  // G * Y  -> A
        float s[16];
#pragma unroll
        for (int cc = 0; cc < 16; ++cc) s[cc] = 0.f;
#pragma unroll
        for (int r8 = 0; r8 < 8; ++r8) {
          int m = r8 + mbase;
          float w = c[r8] * mks[wave][m];
#pragma unroll
          for (int cc = 0; cc < 16; ++cc) s[cc] += w * Ys[wave][m][cc];
        }
#pragma unroll
        for (int cc = 0; cc < 16; ++cc) atomicAdd(&acc_A[cc * DD + d], s[cc]);
      }
    }
    __syncthreads();
  }
  __syncthreads();
  for (int t = tid; t < 3200; t += 256) {
    if (t < DD) o_sb[(size_t)i * DD + t] = acc_sb[t];
    else if (t < 4 * DD) o_v[(size_t)i * 3 * DD + (t - DD)] = acc_v[t - DD];
    else if (t < 9 * DD) o_t[(size_t)i * 5 * DD + (t - 4 * DD)] = acc_t[t - 4 * DD];
    else o_A[(size_t)i * 16 * DD + (t - 9 * DD)] = acc_A[t - 9 * DD];
  }
}

// ---------------------------------------------------------------------------
// K3: attention row kernel. eb-MLP via WMMA (layer2 K=128 chained via LDS
// re-fragmentation), logits -> eb_out, softmax per head, o, wo/LN2/FFN -> dsa
// ---------------------------------------------------------------------------
__global__ __launch_bounds__(256) void k_attn(
    const float* __restrict__ rbf, const float* __restrict__ qmat,
    const float* __restrict__ kmat, const float* __restrict__ vmat,
    const __bf16* __restrict__ fr_ebw1, const __bf16* __restrict__ fr_ebw2,
    const float* __restrict__ eb_b1, const float* __restrict__ eb_b2,
    const float* __restrict__ node_s, const float* __restrict__ wo,
    const float* __restrict__ bo, const float* __restrict__ ln2_g,
    const float* __restrict__ ln2_b, const float* __restrict__ ff1_w,
    const float* __restrict__ ff1_b, const float* __restrict__ ff2_w,
    const float* __restrict__ ff2_b, float* __restrict__ ebout,
    float* __restrict__ o_dsa) {
  __shared__ float logit_s[NN * HH];
  __shared__ __bf16 mid_s[8][32][64];
  __shared__ float q_s[DD];
  __shared__ float o_s[DD];
  __shared__ float sa_s[DD];
  __shared__ float ffa_s[512];
  __shared__ float red[2];

  const int i = blockIdx.x;
  const int tid = threadIdx.x;
  const int lane = tid & 31;
  const int wave = tid >> 5;
  const int mlo = lane & 15;
  const int khi = (lane >> 4) << 3;
  const int mbase = khi;

  if (tid < DD) q_s[tid] = qmat[(size_t)i * DD + tid];
  __syncthreads();

  for (int it = 0; it < 4; ++it) {
    const int tj = wave + it * 8;
    const int j0 = tj * 16;
    v16bf afr;
    {
      const float* r = rbf + (((size_t)i * NN + j0 + mlo) * 32) + khi;
#pragma unroll
      for (int h2 = 0; h2 < 8; ++h2) {
        afr[h2] = (__bf16)r[h2];
        afr[h2 + 8] = (__bf16)r[16 + h2];
      }
    }
    // layer 1: mid = gelu(rbf@eb_w1 + b1), scattered to A-fragment layout
#pragma unroll 1
    for (int t = 0; t < 8; ++t) {
      v16bf bfr = *(const v16bf*)(fr_ebw1 + (((size_t)t * 32 + lane) * 16));
      const int col = t * 16 + mlo;
      const float bias = eb_b1[col];
      v8f c;
#pragma unroll
      for (int r8 = 0; r8 < 8; ++r8) c[r8] = bias;
      c = wmma_bf16(afr, bfr, c);
#pragma unroll
      for (int r8 = 0; r8 < 8; ++r8) {
        float g = gelu_t(c[r8]);
        int m = r8 + mbase;
        int kk_ = col & 31, kc = col >> 5;
        int hi = (kk_ >> 3) & 1;
        int hh = (kk_ & 7) + ((kk_ & 16) ? 8 : 0);
        mid_s[wave][m + (hi << 4)][kc * 16 + hh] = (__bf16)g;
      }
    }
    __syncthreads();
    // layer 2: eb = mid @ eb_w2 (K=128 -> 4 chained WMMAs, N padded to 16)
    v8f c2;
    {
      float b2v = (mlo < HH) ? eb_b2[mlo] : 0.f;
#pragma unroll
      for (int r8 = 0; r8 < 8; ++r8) c2[r8] = b2v;
    }
#pragma unroll 1
    for (int kc = 0; kc < 4; ++kc) {
      v16bf a2 = *(const v16bf*)&mid_s[wave][lane][kc * 16];
      v16bf b2 = *(const v16bf*)(fr_ebw2 + (((size_t)kc * 32 + lane) * 16));
      c2 = wmma_bf16(a2, b2, c2);
    }
    // logits = qk/4 + eb ; also emit eb_out
    if (mlo < HH) {
      const int h = mlo;
#pragma unroll
      for (int r8 = 0; r8 < 8; ++r8) {
        int m = r8 + mbase; int j = j0 + m;
        const float* kr = kmat + (size_t)j * DD + h * 16;
        float dotv = 0.f;
#pragma unroll
        for (int dd2 = 0; dd2 < 16; ++dd2) dotv += q_s[h * 16 + dd2] * kr[dd2];
        float lg = c2[r8] + dotv * 0.25f;
        logit_s[j * HH + h] = lg;
        ebout[((size_t)i * NN + j) * HH + h] = lg;
      }
    }
    __syncthreads();
  }
  __syncthreads();

  // softmax + o : wave w handles head w
  {
    const int h = wave;
    float mx = -1e30f;
    for (int j = lane; j < NN; j += 32) mx = fmaxf(mx, logit_s[j * HH + h]);
#pragma unroll
    for (int off = 16; off >= 1; off >>= 1) mx = fmaxf(mx, __shfl_xor(mx, off, 32));
    float sm = 0.f;
    for (int j = lane; j < NN; j += 32) {
      float p = __expf(logit_s[j * HH + h] - mx);
      logit_s[j * HH + h] = p;
      sm += p;
    }
#pragma unroll
    for (int off = 16; off >= 1; off >>= 1) sm += __shfl_xor(sm, off, 32);
    const float inv = 1.f / sm;
    const int dd2 = lane & 15, half = lane >> 4;
    float acc = 0.f;
    for (int j = half * 256; j < half * 256 + 256; ++j)
      acc += logit_s[j * HH + h] * vmat[(size_t)j * DD + h * 16 + dd2];
    acc += __shfl_xor(acc, 16, 32);
    if (lane < 16) o_s[h * 16 + dd2] = acc * inv;
  }
  __syncthreads();

  // sa = node_s + o@wo + bo
  if (tid < DD) {
    float s = 0.f;
    for (int k2 = 0; k2 < DD; ++k2) s += o_s[k2] * wo[(size_t)k2 * DD + tid];
    sa_s[tid] = node_s[(size_t)i * DD + tid] + s + bo[tid];
  }
  __syncthreads();
  if (tid == 0) { red[0] = 0.f; red[1] = 0.f; }
  __syncthreads();
  if (tid < DD) {
    float x = sa_s[tid];
    atomicAdd(&red[0], x);
    atomicAdd(&red[1], x * x);
  }
  __syncthreads();
  {
    float mean = red[0] * (1.f / DD), var = red[1] * (1.f / DD) - mean * mean;
    float rstd = rsqrtf(var + 1e-5f);
    if (tid < DD) q_s[tid] = (sa_s[tid] - mean) * rstd * ln2_g[tid] + ln2_b[tid];
  }
  __syncthreads();
  for (int cb = 0; cb < 2; ++cb) {
    int col = tid + cb * 256;
    float s = 0.f;
    for (int k2 = 0; k2 < DD; ++k2) s += q_s[k2] * ff1_w[(size_t)k2 * 512 + col];
    ffa_s[col] = gelu_t(s + ff1_b[col]);
  }
  __syncthreads();
  if (tid < DD) {
    float s = 0.f;
    for (int k2 = 0; k2 < 512; ++k2) s += ffa_s[k2] * ff2_w[(size_t)k2 * DD + tid];
    float sa2 = sa_s[tid] + s + ff2_b[tid];
    o_dsa[(size_t)i * DD + tid] = sa2 - node_s[(size_t)i * DD + tid];
  }
}

// ---------------------------------------------------------------------------
// K5: power spectrum + gates + final LN -> h_new, v_new, t_new
// ---------------------------------------------------------------------------
__global__ __launch_bounds__(128) void k_final(
    const float* __restrict__ node_s, const float* __restrict__ node_v,
    const float* __restrict__ node_t, const float* __restrict__ d_sb,
    const float* __restrict__ d_v, const float* __restrict__ d_t,
    const float* __restrict__ d_A, const float* __restrict__ d_dsa,
    const float* __restrict__ vno, const float* __restrict__ tno,
    const float* __restrict__ c_wo, const float* __restrict__ c_bo,
    const float* __restrict__ f_gs_w, const float* __restrict__ f_gs_b,
    const float* __restrict__ f_gv_w, const float* __restrict__ f_gv_b,
    const float* __restrict__ f_gt_w, const float* __restrict__ f_gt_b,
    const float* __restrict__ f_ln_g, const float* __restrict__ f_ln_b,
    const float* __restrict__ scale_v, const float* __restrict__ scale_t,
    float* __restrict__ out_h, float* __restrict__ out_v,
    float* __restrict__ out_t) {
  __shared__ float sl_s[512];
  __shared__ float gin_s[3 * DD];
  __shared__ float red[2];
  const int i = blockIdx.x, d = threadIdx.x;
  const float* Ar = d_A + (size_t)i * 16 * DD + d;
  const float PI4 = 12.566370614359172f;
  { float a = Ar[0]; sl_s[d] = PI4 * a * a; }
  {
    float s = 0.f;
#pragma unroll
    for (int c = 1; c < 4; ++c) { float a = Ar[c * DD]; s += a * a; }
    sl_s[DD + d] = (PI4 / 3.f) * s;
  }
  {
    float s = 0.f;
#pragma unroll
    for (int c = 4; c < 9; ++c) { float a = Ar[c * DD]; s += a * a; }
    sl_s[2 * DD + d] = (PI4 / 5.f) * s;
  }
  {
    float s = 0.f;
#pragma unroll
    for (int c = 9; c < 16; ++c) { float a = Ar[c * DD]; s += a * a; }
    sl_s[3 * DD + d] = (PI4 / 7.f) * s;
  }
  const float dsa = d_dsa[(size_t)i * DD + d];
  const float dsb = d_sb[(size_t)i * DD + d];
  const float vn = vno[(size_t)i * DD + d], tn = tno[(size_t)i * DD + d];
  __syncthreads();
  float dsc = c_bo[d];
  for (int k2 = 0; k2 < 512; ++k2) dsc += sl_s[k2] * c_wo[(size_t)k2 * DD + d];
  gin_s[d] = dsa;
  gin_s[DD + d] = dsb + dsc;
  gin_s[2 * DD + d] = vn + tn;
  __syncthreads();
  float gs = f_gs_b[d];
  for (int k2 = 0; k2 < 3 * DD; ++k2) gs += gin_s[k2] * f_gs_w[(size_t)k2 * DD + d];
  gs = sigm(gs);
  float x = gs * (dsa + dsb + dsc);
  __syncthreads();
  if (d == 0) { red[0] = 0.f; red[1] = 0.f; }
  __syncthreads();
  atomicAdd(&red[0], x);
  atomicAdd(&red[1], x * x);
  __syncthreads();
  float mean = red[0] * (1.f / DD), var = red[1] * (1.f / DD) - mean * mean;
  float hn = node_s[(size_t)i * DD + d] +
             (x - mean) * rsqrtf(var + 1e-5f) * f_ln_g[d] + f_ln_b[d];
  out_h[(size_t)i * DD + d] = hn;
  __syncthreads();
  gin_s[d] = hn;
  gin_s[DD + d] = vn;
  __syncthreads();
  float gv = f_gv_b[d];
  for (int k2 = 0; k2 < 2 * DD; ++k2) gv += gin_s[k2] * f_gv_w[(size_t)k2 * DD + d];
  gv = sigm(gv);
#pragma unroll
  for (int c = 0; c < 3; ++c) {
    size_t off = ((size_t)i * 3 + c) * DD + d;
    out_v[off] = node_v[off] + gv * d_v[off] * scale_v[d];
  }
  __syncthreads();
  gin_s[DD + d] = tn;
  __syncthreads();
  float gt = f_gt_b[d];
  for (int k2 = 0; k2 < 2 * DD; ++k2) gt += gin_s[k2] * f_gt_w[(size_t)k2 * DD + d];
  gt = sigm(gt);
#pragma unroll
  for (int c = 0; c < 5; ++c) {
    size_t off = ((size_t)i * 5 + c) * DD + d;
    out_t[off] = node_t[off] + gt * d_t[off] * scale_t[d];
  }
}

// ---------------------------------------------------------------------------
extern "C" void kernel_launch(void* const* d_in, const int* in_sizes, int n_in,
                              void* d_out, int out_size, void* d_ws,
                              size_t ws_size, hipStream_t stream) {
  (void)in_sizes; (void)n_in; (void)out_size; (void)ws_size;
  const float* node_s = (const float*)d_in[0];
  const float* node_v = (const float*)d_in[1];
  const float* node_t = (const float*)d_in[2];
  const float* rbf    = (const float*)d_in[3];
  const float* r_hat  = (const float*)d_in[4];
  const unsigned char* amask = (const unsigned char*)d_in[5];
  const float* eb_w1 = (const float*)d_in[6];
  const float* eb_b1 = (const float*)d_in[7];
  const float* eb_w2 = (const float*)d_in[8];
  const float* eb_b2 = (const float*)d_in[9];
  const float* ln1_g = (const float*)d_in[10];
  const float* ln1_b = (const float*)d_in[11];
  const float* wq = (const float*)d_in[12];
  const float* bq = (const float*)d_in[13];
  const float* wk = (const float*)d_in[14];
  const float* bk = (const float*)d_in[15];
  const float* wv = (const float*)d_in[16];
  const float* bv = (const float*)d_in[17];
  const float* wo = (const float*)d_in[18];
  const float* bo = (const float*)d_in[19];
  const float* ln2_g = (const float*)d_in[20];
  const float* ln2_b = (const float*)d_in[21];
  const float* ff1_w = (const float*)d_in[22];
  const float* ff1_b = (const float*)d_in[23];
  const float* ff2_w = (const float*)d_in[24];
  const float* ff2_b = (const float*)d_in[25];
  const float* b_w1 = (const float*)d_in[26];
  const float* b_b1 = (const float*)d_in[27];
  const float* b_w2 = (const float*)d_in[28];
  const float* b_b2 = (const float*)d_in[29];
  const float* b_wf = (const float*)d_in[30];
  const float* b_bf = (const float*)d_in[31];
  const float* c_wf = (const float*)d_in[32];
  const float* c_bf = (const float*)d_in[33];
  const float* c_wo = (const float*)d_in[34];
  const float* c_bo = (const float*)d_in[35];
  const float* f_gs_w = (const float*)d_in[36];
  const float* f_gs_b = (const float*)d_in[37];
  const float* f_gv_w = (const float*)d_in[38];
  const float* f_gv_b = (const float*)d_in[39];
  const float* f_gt_w = (const float*)d_in[40];
  const float* f_gt_b = (const float*)d_in[41];
  const float* f_ln_g = (const float*)d_in[42];
  const float* f_ln_b = (const float*)d_in[43];
  const float* scale_v = (const float*)d_in[44];
  const float* scale_t = (const float*)d_in[45];

  float* out = (float*)d_out;
  float* out_h = out;                    // [512,128]
  float* out_v = out + 65536;            // [512,3,128]
  float* out_t = out + 262144;           // [512,5,128]
  float* ebout = out + 589824;           // [512,512,8]

  // workspace layout (floats); total 2374656 floats (~9.5 MB)
  float* ws = (float*)d_ws;
  float* w_q   = ws;
  float* w_k   = ws + 65536;
  float* w_v   = ws + 131072;
  float* w_phi = ws + 196608;   // [512,640]
  float* w_vn  = ws + 524288;
  float* w_tn  = ws + 589824;
  float* w_dsa = ws + 655360;
  float* w_dsb = ws + 720896;
  float* w_dv  = ws + 786432;   // [512,3,128]
  float* w_dt  = ws + 983040;   // [512,5,128]
  float* w_dA  = ws + 1310720;  // [512,16,128]
  __bf16* fr      = (__bf16*)(ws + 2359296);
  __bf16* fr_ebw1 = fr;            // 8 tiles  * 512 = 4096
  __bf16* fr_ebw2 = fr + 4096;     // 4 tiles  = 2048
  __bf16* fr_bwf  = fr + 6144;     // 40 tiles = 20480
  __bf16* fr_cwf  = fr + 26624;    // 8 tiles  = 4096

  k_swz<<<16, 256, 0, stream>>>(eb_w1, fr_ebw1, 32, 128, 128);
  k_swz<<<8, 256, 0, stream>>>(eb_w2, fr_ebw2, 128, 8, 16);
  k_swz<<<80, 256, 0, stream>>>(b_wf, fr_bwf, 32, 640, 640);
  k_swz<<<16, 256, 0, stream>>>(c_wf, fr_cwf, 32, 128, 128);

  k_node<<<512, 128, 0, stream>>>(node_s, node_v, node_t, ln1_g, ln1_b, wq, bq,
                                  wk, bk, wv, bv, b_w1, b_b1, b_w2, b_b2, w_q,
                                  w_k, w_v, w_phi, w_vn, w_tn);

  k_pair<<<512, 256, 0, stream>>>(rbf, r_hat, amask, w_phi, node_v, node_t,
                                  fr_bwf, fr_cwf, b_bf, c_bf, w_dsb, w_dv,
                                  w_dt, w_dA);

  k_attn<<<512, 256, 0, stream>>>(rbf, w_q, w_k, w_v, fr_ebw1, fr_ebw2, eb_b1,
                                  eb_b2, node_s, wo, bo, ln2_g, ln2_b, ff1_w,
                                  ff1_b, ff2_w, ff2_b, ebout, w_dsa);

  k_final<<<512, 128, 0, stream>>>(node_s, node_v, node_t, w_dsb, w_dv, w_dt,
                                   w_dA, w_dsa, w_vn, w_tn, c_wo, c_bo, f_gs_w,
                                   f_gs_b, f_gv_w, f_gv_b, f_gt_w, f_gt_b,
                                   f_ln_g, f_ln_b, scale_v, scale_t, out_h,
                                   out_v, out_t);
}
